// EGNNConv_16458314678921
// MI455X (gfx1250) — matrix operand order repaired and checked by reading, not compiled, not run on hardware
//
#include <hip/hip_runtime.h>

// ---------------- types ----------------
typedef __attribute__((ext_vector_type(16))) __bf16 v16bf;
typedef __attribute__((ext_vector_type(8)))  float  v8f;
typedef __attribute__((ext_vector_type(8)))  unsigned int v8u;
typedef __attribute__((ext_vector_type(4)))  unsigned int v4u;

#define N_NODES 50000
#define N_EDGES 800000
#define FIN 128
#define HID 256
#define FOUT 128

// ---------------- helpers ----------------
__device__ __forceinline__ unsigned short f2bf(float x) {
    unsigned int u = __float_as_uint(x);
    u += 0x7FFFu + ((u >> 16) & 1u);          // round-to-nearest-even
    return (unsigned short)(u >> 16);
}
__device__ __forceinline__ float bf2f(unsigned short h) {
    return __uint_as_float(((unsigned int)h) << 16);
}
__device__ __forceinline__ float siluf(float x) {
    return x / (1.f + __expf(-x));
}
__device__ __forceinline__ void atomAddF(float* p, float v) {
    __hip_atomic_fetch_add(p, v, __ATOMIC_RELAXED, __HIP_MEMORY_SCOPE_AGENT);
}
__device__ __forceinline__ unsigned atomAddU(unsigned* p, unsigned v) {
    return __hip_atomic_fetch_add(p, v, __ATOMIC_RELAXED, __HIP_MEMORY_SCOPE_AGENT);
}
__device__ __forceinline__ v8f wmma_bf16(v16bf a, v16bf b, v8f c) {
    return __builtin_amdgcn_wmma_f32_16x16x32_bf16(false, a, false, b, (short)0, c,
                                                   false, false);
}

// A fragment (16x32 bf16) from row-major bf16 LDS buffer [rows][ldk].
// ISA layout: lanes 0-15 hold K 0-7 & 16-23; lanes 16-31 hold K 8-15 & 24-31.
__device__ __forceinline__ v16bf ldsA(const unsigned short* sm, int ldk, int r0,
                                      int kbase, int lane) {
    int row = r0 + (lane & 15);
    const unsigned short* p = sm + row * ldk + kbase + ((lane >> 4) << 3);
    v4u lo = *(const v4u*)p;          // K: base+half8 + 0..7
    v4u hi = *(const v4u*)(p + 16);   // K: base+16+half8 + 0..7
    v8u u;
    u[0] = lo[0]; u[1] = lo[1]; u[2] = lo[2]; u[3] = lo[3];
    u[4] = hi[0]; u[5] = hi[1]; u[6] = hi[2]; u[7] = hi[3];
    return __builtin_bit_cast(v16bf, u);
}

// B fragment (32x16 bf16) from repacked global weights: 512 bf16 per tile,
// lane-major (lane*16 + i). One coalesced 32B load per lane.
__device__ __forceinline__ v16bf ldgB(const unsigned short* w, int tileIdx, int lane) {
    const v8u* p = (const v8u*)(w + (size_t)tileIdx * 512) + lane;
    return __builtin_bit_cast(v16bf, *p);
}

// ---------------- prep kernels ----------------
__global__ void zero_f32(float* __restrict__ p, long n) {
    long i = (long)blockIdx.x * blockDim.x + threadIdx.x;
    long stride = (long)gridDim.x * blockDim.x;
    for (; i < n; i += stride) p[i] = 0.f;
}

// Repack f32 weight [K x N] row-major -> bf16 B-fragment tiles.
__global__ void repack_wt(const float* __restrict__ W, unsigned short* __restrict__ out,
                          int N) {
    int ntiles = N >> 4;
    int kt = blockIdx.x / ntiles;
    int nt = blockIdx.x % ntiles;
    int t = threadIdx.x;          // 512 threads = 32 lanes x 16 elems
    int lane = t >> 4, i = t & 15;
    int k   = kt * 32 + ((lane >> 4) << 3) + (i & 7) + ((i >> 3) << 4);
    int col = nt * 16 + (lane & 15);
    out[(size_t)blockIdx.x * 512 + lane * 16 + i] = f2bf(W[(size_t)k * N + col]);
}

// ---------------- CSR build (replaces 205M float atomics with streaming) ----------------
__global__ void hist_dst(const int* __restrict__ dst, unsigned* __restrict__ cnt) {
    long i = (long)blockIdx.x * blockDim.x + threadIdx.x;
    if (i < N_EDGES) atomAddU(&cnt[dst[i]], 1u);
}

// Single-block exclusive scan over 50000 counts -> offsets[50001], cursor copy.
__global__ __launch_bounds__(1024) void scan_offsets(const unsigned* __restrict__ cnt,
                                                     unsigned* __restrict__ offs,
                                                     unsigned* __restrict__ cursor,
                                                     int n) {
    __shared__ unsigned sm[1024];
    __shared__ unsigned carry;
    if (threadIdx.x == 0) carry = 0;
    __syncthreads();
    for (int base = 0; base < n; base += 1024) {
        int i = base + threadIdx.x;
        unsigned v = (i < n) ? cnt[i] : 0u;
        sm[threadIdx.x] = v;
        __syncthreads();
        for (int o = 1; o < 1024; o <<= 1) {
            unsigned t = (threadIdx.x >= (unsigned)o) ? sm[threadIdx.x - o] : 0u;
            __syncthreads();
            sm[threadIdx.x] += t;
            __syncthreads();
        }
        unsigned excl = sm[threadIdx.x] - v;
        if (i < n) { offs[i] = carry + excl; cursor[i] = carry + excl; }
        __syncthreads();
        if (threadIdx.x == 1023) carry += sm[1023];
        __syncthreads();
    }
    if (threadIdx.x == 0) offs[n] = carry;
}

__global__ void scatter_edges(const int* __restrict__ dst, unsigned* __restrict__ cursor,
                              unsigned* __restrict__ eidx) {
    long i = (long)blockIdx.x * blockDim.x + threadIdx.x;
    if (i < N_EDGES) {
        unsigned p = atomAddU(&cursor[dst[i]], 1u);
        eidx[p] = (unsigned)i;
    }
}

// ---------------- fused edge pipeline ----------------
// 64 edges/block. Three chained GEMMs through one in-place LDS buffer.
// USE_CSR: write msg_h/msg_x to global buffers (no float atomics).
// !USE_CSR: legacy float-atomic scatter fallback (small-workspace case).
template <bool USE_CSR>
__global__ __launch_bounds__(256) void egnn_edge(
    const float* __restrict__ node_feat, const float* __restrict__ coord,
    const int* __restrict__ src, const int* __restrict__ dst,
    const unsigned short* __restrict__ We1p, const float* __restrict__ We1raw,
    const float* __restrict__ be1,
    const unsigned short* __restrict__ We2p, const float* __restrict__ be2,
    const unsigned short* __restrict__ Wc1p, const float* __restrict__ bc1,
    const float* __restrict__ Wc2,
    unsigned short* __restrict__ msgH, float* __restrict__ msgX,
    float* __restrict__ h_neigh, float* __restrict__ x_sum, float* __restrict__ deg) {
    __shared__ __align__(16) unsigned short smF[64 * 256];   // 32 KB, reused per layer
    __shared__ float s_rad[64];
    __shared__ float s_xd[64 * 3];
    __shared__ int   s_dst[64];

    const int tid = threadIdx.x, lane = tid & 31, wave = tid >> 5;
    const long eBase = (long)blockIdx.x * 64;

    // ---- gather concat(feat[src], feat[dst]) as bf16 + geometry ----
    {
        int e = tid >> 2, sub = tid & 3;           // 4 threads / edge
        long eg = eBase + e;
        int s = src[eg], d = dst[eg];
        if (sub == 0) {
            s_dst[e] = d;
            float dx = coord[(size_t)s * 3 + 0] - coord[(size_t)d * 3 + 0];
            float dy = coord[(size_t)s * 3 + 1] - coord[(size_t)d * 3 + 1];
            float dz = coord[(size_t)s * 3 + 2] - coord[(size_t)d * 3 + 2];
            float r2 = dx * dx + dy * dy + dz * dz;
            float inv = 1.f / (sqrtf(r2) + 1e-30f);
            s_rad[e] = r2;
            s_xd[e * 3 + 0] = dx * inv;
            s_xd[e * 3 + 1] = dy * inv;
            s_xd[e * 3 + 2] = dz * inv;
        }
        const float* row = node_feat + (size_t)(sub < 2 ? s : d) * FIN + ((sub & 1) << 6);
        unsigned short* o = smF + e * 256 + sub * 64;
#pragma unroll
        for (int i = 0; i < 64; i += 4) {
            float4 v = *(const float4*)(row + i);
            o[i + 0] = f2bf(v.x); o[i + 1] = f2bf(v.y);
            o[i + 2] = f2bf(v.z); o[i + 3] = f2bf(v.w);
        }
    }
    __syncthreads();

    const int mt = wave & 3, nh = wave >> 2;       // 4 M-tiles, 2 N-halves
    const int cn = lane & 15, chi = (lane >> 4) << 3;
    const int r0 = mt * 16;
    v8f acc[8];

    // ---- layer 1: h1 = silu(f @ We1[:256] + radial*We1[256] + be1) ----
#pragma unroll
    for (int j = 0; j < 8; j++) acc[j] = (v8f){0, 0, 0, 0, 0, 0, 0, 0};
    for (int kt = 0; kt < 8; kt++) {
        v16bf a = ldsA(smF, 256, r0, kt * 32, lane);
#pragma unroll
        for (int j = 0; j < 8; j++) {
            v16bf b = ldgB(We1p, kt * 16 + nh * 8 + j, lane);
            acc[j] = wmma_bf16(a, b, acc[j]);
        }
    }
    __syncthreads();   // all reads of smF done -> safe to overwrite
#pragma unroll
    for (int j = 0; j < 8; j++) {
        int col = (nh * 8 + j) * 16 + cn;
        float bb = be1[col], w256 = We1raw[256 * HID + col];
#pragma unroll
        for (int r = 0; r < 8; r++) {
            int m = r0 + chi + r;
            float x = acc[j][r] + bb + s_rad[m] * w256;
            smF[m * 256 + col] = f2bf(siluf(x));
        }
    }
    __syncthreads();

    // ---- layer 2: msg_h = silu(h1 @ We2 + be2) ----
#pragma unroll
    for (int j = 0; j < 8; j++) acc[j] = (v8f){0, 0, 0, 0, 0, 0, 0, 0};
    for (int kt = 0; kt < 8; kt++) {
        v16bf a = ldsA(smF, 256, r0, kt * 32, lane);
#pragma unroll
        for (int j = 0; j < 8; j++) {
            v16bf b = ldgB(We2p, kt * 16 + nh * 8 + j, lane);
            acc[j] = wmma_bf16(a, b, acc[j]);
        }
    }
    __syncthreads();
#pragma unroll
    for (int j = 0; j < 8; j++) {
        int col = (nh * 8 + j) * 16 + cn;
        float bb = be2[col];
#pragma unroll
        for (int r = 0; r < 8; r++) {
            int m = r0 + chi + r;
            float x = siluf(acc[j][r] + bb);
            smF[m * 256 + col] = f2bf(x);
            if constexpr (!USE_CSR)
                atomAddF(&h_neigh[(size_t)s_dst[m] * HID + col], x);
        }
    }
    __syncthreads();

    if constexpr (USE_CSR) {
        // cooperative coalesced b128 copy: msg_h tile -> global bf16 buffer
        const uint4* sp = (const uint4*)smF;
        uint4* dp = (uint4*)(msgH + (size_t)eBase * 256);
#pragma unroll
        for (int i = tid; i < 2048; i += 256) dp[i] = sp[i];
    }

    // ---- layer 3: c = silu(msg_h @ Wc1 + bc1) ----
#pragma unroll
    for (int j = 0; j < 8; j++) acc[j] = (v8f){0, 0, 0, 0, 0, 0, 0, 0};
    for (int kt = 0; kt < 8; kt++) {
        v16bf a = ldsA(smF, 256, r0, kt * 32, lane);
#pragma unroll
        for (int j = 0; j < 8; j++) {
            v16bf b = ldgB(Wc1p, kt * 16 + nh * 8 + j, lane);
            acc[j] = wmma_bf16(a, b, acc[j]);
        }
    }
    __syncthreads();
#pragma unroll
    for (int j = 0; j < 8; j++) {
        int col = (nh * 8 + j) * 16 + cn;
        float bb = bc1[col];
#pragma unroll
        for (int r = 0; r < 8; r++) {
            int m = r0 + chi + r;
            smF[m * 256 + col] = f2bf(siluf(acc[j][r] + bb));
        }
    }
    __syncthreads();

    // ---- coord branch: s = c . Wc2 ; msg_x = s * x_diff ----
    {
        int e = tid >> 2, sub = tid & 3;
        const unsigned short* crow = smF + e * 256 + sub * 64;
        const float* wr = Wc2 + sub * 64;
        float sum = 0.f;
#pragma unroll
        for (int i = 0; i < 64; i++) sum += bf2f(crow[i]) * wr[i];
        sum += __shfl_xor(sum, 1, 32);
        sum += __shfl_xor(sum, 2, 32);
        if (sub == 0) {
            long eg = eBase + e;
            if constexpr (USE_CSR) {
                msgX[(size_t)eg * 4 + 0] = sum * s_xd[e * 3 + 0];
                msgX[(size_t)eg * 4 + 1] = sum * s_xd[e * 3 + 1];
                msgX[(size_t)eg * 4 + 2] = sum * s_xd[e * 3 + 2];
            } else {
                int dn = s_dst[e];
                atomAddF(&deg[dn], 1.f);
                atomAddF(&x_sum[(size_t)dn * 3 + 0], sum * s_xd[e * 3 + 0]);
                atomAddF(&x_sum[(size_t)dn * 3 + 1], sum * s_xd[e * 3 + 1]);
                atomAddF(&x_sum[(size_t)dn * 3 + 2], sum * s_xd[e * 3 + 2]);
            }
        }
    }
}

// ---------------- CSR aggregation: h_neigh (sum) + x output (mean) ----------------
__global__ __launch_bounds__(256) void aggregate(
    const unsigned short* __restrict__ msgH, const float* __restrict__ msgX,
    const unsigned* __restrict__ offs, const unsigned* __restrict__ eidx,
    const float* __restrict__ coord, float* __restrict__ h_neigh,
    float* __restrict__ xout) {
    __shared__ unsigned se[64];
    const int n = blockIdx.x, tid = threadIdx.x;
    const unsigned beg = offs[n], end = offs[n + 1];
    float s = 0.f, sx = 0.f;
    for (unsigned base = beg; base < end; base += 64) {
        unsigned cnt = end - base; if (cnt > 64) cnt = 64;
        if ((unsigned)tid < cnt) se[tid] = eidx[base + tid];
        __syncthreads();
        for (unsigned j = 0; j < cnt; j++) {
            size_t e = se[j];
            s += bf2f(msgH[e * 256 + tid]);          // coalesced across 256 lanes
            if (tid < 3) sx += msgX[e * 4 + tid];
        }
        __syncthreads();
    }
    h_neigh[(size_t)n * HID + tid] = s;
    if (tid < 3) {
        float d = fmaxf((float)(end - beg), 1.f);
        xout[(size_t)n * 3 + tid] = coord[(size_t)n * 3 + tid] + sx / d;
    }
}

// ---------------- node MLP ----------------
__global__ __launch_bounds__(256) void egnn_node(
    const float* __restrict__ node_feat, const float* __restrict__ h_neigh,
    const unsigned short* __restrict__ Wn1p, const float* __restrict__ bn1,
    const unsigned short* __restrict__ Wn2p, const float* __restrict__ bn2,
    float* __restrict__ h_pre) {
    __shared__ __align__(16) unsigned short smA[32 * 384];   // 24 KB
    __shared__ __align__(16) unsigned short smM[32 * 256];   // 16 KB
    const int tid = threadIdx.x, lane = tid & 31, wave = tid >> 5;
    const int nodeBase = blockIdx.x * 32;

    for (int idx = tid; idx < 32 * 384; idx += 256) {
        int r = idx / 384, c = idx - r * 384;
        int n = nodeBase + r; if (n >= N_NODES) n = N_NODES - 1;
        float v = (c < FIN) ? node_feat[(size_t)n * FIN + c]
                            : h_neigh[(size_t)n * HID + (c - FIN)];
        smA[idx] = f2bf(v);
    }
    __syncthreads();

    const int cn = lane & 15, chi = (lane >> 4) << 3;

    // layer 1: [32x384] x [384x256]
    {
        const int mt = wave & 1, nq = wave >> 1;   // 2 M-tiles, 4 N-quarters
        v8f acc[4];
#pragma unroll
        for (int j = 0; j < 4; j++) acc[j] = (v8f){0, 0, 0, 0, 0, 0, 0, 0};
        for (int kt = 0; kt < 12; kt++) {
            v16bf a = ldsA(smA, 384, mt * 16, kt * 32, lane);
#pragma unroll
            for (int j = 0; j < 4; j++) {
                v16bf b = ldgB(Wn1p, kt * 16 + nq * 4 + j, lane);
                acc[j] = wmma_bf16(a, b, acc[j]);
            }
        }
        __syncthreads();
#pragma unroll
        for (int j = 0; j < 4; j++) {
            int col = (nq * 4 + j) * 16 + cn;
            float bb = bn1[col];
#pragma unroll
            for (int r = 0; r < 8; r++) {
                int m = mt * 16 + chi + r;
                smM[m * 256 + col] = f2bf(siluf(acc[j][r] + bb));
            }
        }
        __syncthreads();
    }
    // layer 2: [32x256] x [256x128]
    {
        const int mt = wave & 1, nq = wave >> 1;
        v8f acc[2];
#pragma unroll
        for (int j = 0; j < 2; j++) acc[j] = (v8f){0, 0, 0, 0, 0, 0, 0, 0};
        for (int kt = 0; kt < 8; kt++) {
            v16bf a = ldsA(smM, 256, mt * 16, kt * 32, lane);
#pragma unroll
            for (int j = 0; j < 2; j++) {
                v16bf b = ldgB(Wn2p, kt * 8 + nq * 2 + j, lane);   // ntiles = 8
                acc[j] = wmma_bf16(a, b, acc[j]);
            }
        }
#pragma unroll
        for (int j = 0; j < 2; j++) {
            int col = (nq * 2 + j) * 16 + cn;
            float bb = bn2[col];
#pragma unroll
            for (int r = 0; r < 8; r++) {
                int m = mt * 16 + chi + r;
                int node = nodeBase + m;
                if (node < N_NODES) h_pre[(size_t)node * FOUT + col] = acc[j][r] + bb;
            }
        }
    }
}

// ---------------- epilogue kernels ----------------
__global__ void egnn_coord(const float* __restrict__ coord, const float* __restrict__ x_sum,
                           const float* __restrict__ deg, float* __restrict__ xout) {
    int i = blockIdx.x * 256 + threadIdx.x;
    if (i < N_NODES) {
        float d = fmaxf(deg[i], 1.f);
#pragma unroll
        for (int c = 0; c < 3; c++)
            xout[(size_t)i * 3 + c] = coord[(size_t)i * 3 + c] + x_sum[(size_t)i * 3 + c] / d;
    }
}

// Coalesced BN partial sums: each block covers 250 rows x 128 cols.
__global__ __launch_bounds__(256) void bn_partial(const float* __restrict__ h_pre,
                                                  float* __restrict__ gsum,
                                                  float* __restrict__ gsq) {
    __shared__ float ss[256], sq[256];
    int col = threadIdx.x & 127, half = threadIdx.x >> 7;
    long r0 = (long)blockIdx.x * 250;
    float s = 0.f, q = 0.f;
    for (long r = r0 + half; r < r0 + 250 && r < N_NODES; r += 2) {
        float v = h_pre[r * FOUT + col];
        s += v; q += v * v;
    }
    ss[threadIdx.x] = s; sq[threadIdx.x] = q;
    __syncthreads();
    if (half == 0) {
        atomAddF(&gsum[col], ss[col] + ss[col + 128]);
        atomAddF(&gsq[col],  sq[col] + sq[col + 128]);
    }
}

__global__ void bn_finalize(const float* __restrict__ gsum, const float* __restrict__ gsq,
                            float* __restrict__ mu, float* __restrict__ rstd) {
    int c = threadIdx.x;   // 128 threads
    float m = gsum[c] / (float)N_NODES;
    float var = gsq[c] / (float)N_NODES - m * m;
    mu[c] = m;
    rstd[c] = rsqrtf(var + 1e-5f);
}

__global__ void bn_apply(const float* __restrict__ h_pre, const float* __restrict__ mu,
                         const float* __restrict__ rstd, const float* __restrict__ g,
                         const float* __restrict__ b, float* __restrict__ hout) {
    long i = (long)blockIdx.x * 256 + threadIdx.x;
    if (i < (long)N_NODES * FOUT) {
        int c = (int)(i & (FOUT - 1));
        hout[i] = (h_pre[i] - mu[c]) * rstd[c] * g[c] + b[c];
    }
}

// ---------------- launcher ----------------
static inline char* alignup(char* p) {
    return (char*)(((size_t)p + 255) & ~(size_t)255);
}

extern "C" void kernel_launch(void* const* d_in, const int* in_sizes, int n_in,
                              void* d_out, int out_size, void* d_ws, size_t ws_size,
                              hipStream_t stream) {
    const float* node_feat = (const float*)d_in[0];
    const float* coord     = (const float*)d_in[1];
    const int*   src       = (const int*)d_in[2];
    const int*   dst       = (const int*)d_in[3];
    const float* We1  = (const float*)d_in[4];
    const float* be1  = (const float*)d_in[5];
    const float* We2  = (const float*)d_in[6];
    const float* be2  = (const float*)d_in[7];
    const float* Wc1  = (const float*)d_in[8];
    const float* bc1  = (const float*)d_in[9];
    const float* Wc2  = (const float*)d_in[10];
    const float* Wn1  = (const float*)d_in[11];
    const float* bn1  = (const float*)d_in[12];
    const float* Wn2  = (const float*)d_in[13];
    const float* bn2  = (const float*)d_in[14];
    const float* gmm  = (const float*)d_in[15];
    const float* bta  = (const float*)d_in[16];

    // ---- workspace carve-up ----
    char* w = (char*)d_ws;
    unsigned short* We1p = (unsigned short*)w; w += (size_t)8  * 16 * 512 * 2;
    unsigned short* We2p = (unsigned short*)w; w += (size_t)8  * 16 * 512 * 2;
    unsigned short* Wc1p = (unsigned short*)w; w += (size_t)8  * 16 * 512 * 2;
    unsigned short* Wn1p = (unsigned short*)w; w += (size_t)12 * 16 * 512 * 2;
    unsigned short* Wn2p = (unsigned short*)w; w += (size_t)8  * 8  * 512 * 2;
    float* h_pre = (float*)w; w += (size_t)N_NODES * FOUT * 4;
    float* mu    = (float*)w; w += 128 * 4;
    float* rstd  = (float*)w; w += 128 * 4;
    unsigned* offs   = (unsigned*)w; w += (size_t)(N_NODES + 1) * 4;
    unsigned* cursor = (unsigned*)w; w += (size_t)N_NODES * 4;
    w = alignup(w);
    // zeroed region (contiguous): h_neigh | x_sum | deg | cnt | gsum | gsq
    float*    zbase   = (float*)w;
    float*    h_neigh = (float*)w;    w += (size_t)N_NODES * HID * 4;
    float*    x_sum   = (float*)w;    w += (size_t)N_NODES * 3 * 4;
    float*    deg     = (float*)w;    w += (size_t)N_NODES * 4;
    unsigned* cnt     = (unsigned*)w; w += (size_t)N_NODES * 4;
    float*    gsum    = (float*)w;    w += 128 * 4;
    float*    gsq     = (float*)w;    w += 128 * 4;
    long nzero = ((float*)w) - zbase;
    // CSR extras
    w = alignup(w);
    unsigned*       eidx = (unsigned*)w;       w += (size_t)N_EDGES * 4;
    float*          msgX = (float*)w;          w += (size_t)N_EDGES * 4 * 4;
    w = alignup(w);
    unsigned short* msgH = (unsigned short*)w; w += (size_t)N_EDGES * HID * 2;
    const bool use_csr = ((size_t)(w - (char*)d_ws) <= ws_size);

    float* hout = (float*)d_out;                       // [50000 x 128]
    float* xout = hout + (size_t)N_NODES * FOUT;       // [50000 x 3]

    // zero accumulators (incl. CSR histogram + BN partial sums)
    zero_f32<<<4096, 256, 0, stream>>>(zbase, nzero);

    // repack weights to bf16 WMMA B-fragment layout
    repack_wt<<<8 * 16, 512, 0, stream>>>(We1, We1p, 256);
    repack_wt<<<8 * 16, 512, 0, stream>>>(We2, We2p, 256);
    repack_wt<<<8 * 16, 512, 0, stream>>>(Wc1, Wc1p, 256);
    repack_wt<<<12 * 16, 512, 0, stream>>>(Wn1, Wn1p, 256);
    repack_wt<<<8 * 8, 512, 0, stream>>>(Wn2, Wn2p, 128);

    if (use_csr) {
        // CSR by destination
        hist_dst<<<(N_EDGES + 255) / 256, 256, 0, stream>>>(dst, cnt);
        scan_offsets<<<1, 1024, 0, stream>>>(cnt, offs, cursor, N_NODES);
        scatter_edges<<<(N_EDGES + 255) / 256, 256, 0, stream>>>(dst, cursor, eidx);

        // fused per-edge MLP chain, streaming msg_h / msg_x
        egnn_edge<true><<<N_EDGES / 64, 256, 0, stream>>>(
            node_feat, coord, src, dst, We1p, We1, be1, We2p, be2, Wc1p, bc1, Wc2,
            msgH, msgX, h_neigh, x_sum, deg);

        // gather-based aggregation (also writes final x output)
        aggregate<<<N_NODES, 256, 0, stream>>>(msgH, msgX, offs, eidx, coord,
                                               h_neigh, xout);
    } else {
        // atomic-scatter fallback for small workspaces
        egnn_edge<false><<<N_EDGES / 64, 256, 0, stream>>>(
            node_feat, coord, src, dst, We1p, We1, be1, We2p, be2, Wc1p, bc1, Wc2,
            nullptr, nullptr, h_neigh, x_sum, deg);
        egnn_coord<<<(N_NODES + 255) / 256, 256, 0, stream>>>(coord, x_sum, deg, xout);
    }

    // node MLP
    egnn_node<<<(N_NODES + 31) / 32, 256, 0, stream>>>(
        node_feat, h_neigh, Wn1p, bn1, Wn2p, bn2, h_pre);

    // batch-norm (training stats, biased var)
    bn_partial<<<200, 256, 0, stream>>>(h_pre, gsum, gsq);
    bn_finalize<<<1, 128, 0, stream>>>(gsum, gsq, mu, rstd);
    bn_apply<<<(int)(((long)N_NODES * FOUT + 255) / 256), 256, 0, stream>>>(
        h_pre, mu, rstd, gmm, bta, hout);
}
